// TopkSparseAutoEncoder_v2_7456063225989
// MI455X (gfx1250) — matrix operand order repaired and checked by reading, not compiled, not run on hardware
//
#include <hip/hip_runtime.h>
#include <hip/hip_bf16.h>

// Topk (k=3) Sparse AutoEncoder, fused for MI455X (gfx1250, wave32).
//
// Pre-pass: W_enc f32 -> bf16 copy in scratch (~24 MB, one HBM sweep).
// Encode:   bf16 WMMA GEMM (f32 accum); A fragments register-resident
//           (afrag[24], 192 VGPRs/lane); W_enc panels streamed to a 4-deep
//           LDS ring by the Tensor Data Mover (3 loads in flight,
//           s_wait_tensorcnt 2 steady-state); fused running per-row top-3.
// Then: live-mask zero/reduce -> count, and sparse 3-column gather decode.

#define BATCH      16384
#define MODEL_DIM  768
#define SAE_HIDDEN 16384
#define KTOP       3

#define MB     128            // batch rows per block (8 waves x 16 rows)
#define WAVES  8
#define TPB    256
#define APAD   8
#define ASTRIDE (MODEL_DIM + APAD)   // 776 ushorts, 16B-aligned rows
// B panel in LDS: TDM pads 16 B after every 512 B -> 1536+48 = 1584 B rows.
#define BSTRIDE 792                  // ushorts per padded B row
#define BUFB    (32 * BSTRIDE * 2)   // bytes per ring buffer (50,688)

typedef __attribute__((ext_vector_type(16))) __bf16 v16bf;
typedef __attribute__((ext_vector_type(8)))  float  v8f;
typedef __attribute__((ext_vector_type(4)))  unsigned int u32x4;
typedef __attribute__((ext_vector_type(8)))  int i32x8;
typedef __attribute__((ext_vector_type(4)))  int i32x4;

static __device__ __forceinline__ unsigned short f2bf(float f) {
  unsigned int u = __float_as_uint(f);
  u += 0x7FFFu + ((u >> 16) & 1u);        // round-to-nearest-even
  return (unsigned short)(u >> 16);
}

static __device__ __forceinline__ v16bf ld_frag(const unsigned short* p0,
                                                const unsigned short* p1) {
  union { uint4 q[2]; v16bf v; } r;
  r.q[0] = *reinterpret_cast<const uint4*>(p0);   // ds_load_b128
  r.q[1] = *reinterpret_cast<const uint4*>(p1);   // ds_load_b128
  return r.v;
}

// ---- Tensor Data Mover: load one 32-row x 768-col bf16 W_enc panel to LDS.
// D# per CDNA5 ISA ch.8: 2D tile, data_size=2B, pad 4 DWORDs per 128 DWORDs.
static __device__ __forceinline__ void tdm_load_panel(
    const unsigned short* gsrc, unsigned lds_off) {
  unsigned long long ga = (unsigned long long)(size_t)gsrc;
  u32x4 g0;
  g0.x = 1u;                                           // count=1, no gather
  g0.y = lds_off;                                      // lds_addr
  g0.z = (unsigned)(ga & 0xFFFFFFFFull);               // global_addr[31:0]
  g0.w = (unsigned)((ga >> 32) & 0x01FFFFFFull)        // global_addr[56:32]
       | 0x80000000u;                                  // type=2 ("image")
  i32x8 g1;
  g1[0] = (int)((1u << 16)     // data_size = 2 bytes
              | (1u << 20)     // pad_enable
              | (6u << 22)     // pad_interval: 128 DWORDs (512 B)
              | (3u << 25));   // pad_amount:   4 DWORDs  (16 B)
  g1[1] = (int)(768u << 16);                 // tensor_dim0 = 768 (lo16)
  g1[2] = (int)(16384u << 16);               // dim0 hi16=0 | tensor_dim1 lo16
  g1[3] = (int)(768u << 16);                 // dim1 hi16=0 | tile_dim0 = 768
  g1[4] = 32;                                // tile_dim1 = 32, tile_dim2 = 0
  g1[5] = 768;                               // tensor_dim0_stride lo32
  g1[6] = 0;                                 // stride0 hi16 | stride1 lo16
  g1[7] = 0;                                 // stride1 hi32
  i32x4 z4 = {0, 0, 0, 0};
#if defined(__clang_major__) && (__clang_major__ >= 23)
  i32x8 z8 = {0, 0, 0, 0, 0, 0, 0, 0};
  __builtin_amdgcn_tensor_load_to_lds(g0, g1, z4, z4, z8, 0);
#else
  __builtin_amdgcn_tensor_load_to_lds(g0, g1, z4, z4, 0);
#endif
}

struct MergeBuf {
  float cval[WAVES][32][8][KTOP];
  int   cidx[WAVES][32][8][KTOP];
};
// Phased LDS reuse: A staging -> 4-deep B ring -> merge scratch.
union ShareAll {
  unsigned short a[MB * ASTRIDE];     // phase 1: 198,656 B
  unsigned short b[4][32 * BSTRIDE];  // phase 2: 202,752 B
  MergeBuf m;                         // phase 3:  49,152 B
};

__global__ void sae_zero_live(int* __restrict__ live) {
  int i = blockIdx.x * blockDim.x + threadIdx.x;
  if (i < SAE_HIDDEN) live[i] = 0;
}

__global__ void sae_cvt_bf16(const float* __restrict__ src,
                             unsigned short* __restrict__ dst, int n) {
  int i = (blockIdx.x * 256 + threadIdx.x) * 4;
  if (i < n) {
    const float4 f = *reinterpret_cast<const float4*>(src + i);
    unsigned long long p = (unsigned long long)f2bf(f.x)
                         | ((unsigned long long)f2bf(f.y) << 16)
                         | ((unsigned long long)f2bf(f.z) << 32)
                         | ((unsigned long long)f2bf(f.w) << 48);
    *reinterpret_cast<unsigned long long*>(dst + i) = p;
  }
}

// ---------------- main path: TDM-staged bf16 panels ----------------
__global__ __launch_bounds__(TPB, 1)
void sae_encode_topk_tdm(const float* __restrict__ X,             // [B, D]
                         const unsigned short* __restrict__ Wb,   // [H, D] bf16
                         const float* __restrict__ benc,          // [H]
                         float* __restrict__ topv, int* __restrict__ topi,
                         int* __restrict__ live)
{
  __shared__ alignas(16) ShareAll sh;

  const int tid     = threadIdx.x;
  const int wave    = tid >> 5;
  const int lane    = tid & 31;
  const int rowBase = blockIdx.x * MB;

  // ---- phase 1: stage A tile (128 x 768) f32 -> bf16 in LDS ----
  for (int e = tid * 4; e < MB * MODEL_DIM; e += TPB * 4) {
    int m = e / MODEL_DIM;
    int k = e - m * MODEL_DIM;
    const float4 f = *reinterpret_cast<const float4*>(
        X + (size_t)(rowBase + m) * MODEL_DIM + k);
    unsigned long long p = (unsigned long long)f2bf(f.x)
                         | ((unsigned long long)f2bf(f.y) << 16)
                         | ((unsigned long long)f2bf(f.z) << 32)
                         | ((unsigned long long)f2bf(f.w) << 48);
    *reinterpret_cast<unsigned long long*>(&sh.a[m * ASTRIDE + k]) = p;
  }
  __syncthreads();

  // ---- phase 2: extract this wave's A fragments to registers ----
  const int half = lane >> 4;
  const int ha   = half ? 8 : 0;            // A chunk select (ISA A layout)
  const int hb0  = half ? 8 : 0;            // B chunk selects (ISA B layout)
  const int hb1  = half ? 24 : 16;
  const int bcol = lane & 15;
  const unsigned short* arow = &sh.a[((wave << 4) + (lane & 15)) * ASTRIDE];

  v16bf afrag[24];                          // 16x768 bf16 / 32 lanes = 192 VGPR
#pragma unroll
  for (int kk = 0; kk < 24; ++kk)
    afrag[kk] = ld_frag(arow + kk * 32 + ha, arow + kk * 32 + 16 + ha);
  __syncthreads();                          // A LDS dead; ring may be written

  float tv[8][KTOP]; int ti[8][KTOP];
#pragma unroll
  for (int i = 0; i < 8; ++i) {
    tv[i][0] = tv[i][1] = tv[i][2] = -3.4e38f;
    ti[i][0] = ti[i][1] = ti[i][2] = 0;
  }

  // ---- phase 3: TDM ring (4 buffers, 3 panels in flight) ----
  const unsigned ldsb = (unsigned)(size_t)&sh.b[0][0];
  if (wave == 0) {
    tdm_load_panel(Wb,                                ldsb);
    tdm_load_panel(Wb + (size_t)1 * 32 * MODEL_DIM,   ldsb + 1 * BUFB);
    tdm_load_panel(Wb + (size_t)2 * 32 * MODEL_DIM,   ldsb + 2 * BUFB);
  }

  const int NT = SAE_HIDDEN / 32;   // 512 double-tiles (32 columns each)
  for (int t = 0; t < NT; ++t) {
    // TDM completes in-order: <=2 outstanding => panel t has landed.
    if (t < NT - 2)       __builtin_amdgcn_s_wait_tensorcnt(2);
    else if (t == NT - 2) __builtin_amdgcn_s_wait_tensorcnt(1);
    else                  __builtin_amdgcn_s_wait_tensorcnt(0);
    __syncthreads();                        // visible to all waves
    if (wave == 0 && t + 3 < NT)
      tdm_load_panel(Wb + (size_t)(t + 3) * 32 * MODEL_DIM,
                     ldsb + (unsigned)((t + 3) & 3) * BUFB);

    const unsigned short* bt = &sh.b[t & 3][0];
    const unsigned short* p0a = bt + bcol * BSTRIDE + hb0;  // cols t*32+bcol
    const unsigned short* p0b = bt + bcol * BSTRIDE + hb1;
    const unsigned short* p1a = p0a + 16 * BSTRIDE;         // cols +16
    const unsigned short* p1b = p0b + 16 * BSTRIDE;

    v8f c0 = {0.f,0.f,0.f,0.f,0.f,0.f,0.f,0.f};
    v8f c1 = {0.f,0.f,0.f,0.f,0.f,0.f,0.f,0.f};
#pragma unroll
    for (int kk = 0; kk < 24; ++kk) {
      const int k0 = kk * 32;
      const int kp = k0 + 8 * (k0 >> 8);    // TDM pad: +8 ushorts per 256 elems
      v16bf b0 = ld_frag(p0a + kp, p0b + kp);
      v16bf b1 = ld_frag(p1a + kp, p1b + kp);
      c0 = __builtin_amdgcn_wmma_f32_16x16x32_bf16(
               false, afrag[kk], false, b0, (short)0, c0, false, false);
      c1 = __builtin_amdgcn_wmma_f32_16x16x32_bf16(
               false, afrag[kk], false, b1, (short)0, c1, false, false);
    }

    const int n0 = t * 32 + bcol;           // both cols === bcol (mod 16)
    const float bia0 = benc[n0];
    const float bia1 = benc[n0 + 16];
#pragma unroll
    for (int i = 0; i < 8; ++i) {
      float v = c0[i] + bia0;
      if (v > tv[i][0]) { tv[i][2]=tv[i][1]; ti[i][2]=ti[i][1];
                          tv[i][1]=tv[i][0]; ti[i][1]=ti[i][0];
                          tv[i][0]=v;        ti[i][0]=n0; }
      else if (v > tv[i][1]) { tv[i][2]=tv[i][1]; ti[i][2]=ti[i][1];
                               tv[i][1]=v; ti[i][1]=n0; }
      else if (v > tv[i][2]) { tv[i][2]=v; ti[i][2]=n0; }
      float w = c1[i] + bia1;
      if (w > tv[i][0]) { tv[i][2]=tv[i][1]; ti[i][2]=ti[i][1];
                          tv[i][1]=tv[i][0]; ti[i][1]=ti[i][0];
                          tv[i][0]=w;        ti[i][0]=n0+16; }
      else if (w > tv[i][1]) { tv[i][2]=tv[i][1]; ti[i][2]=ti[i][1];
                               tv[i][1]=w; ti[i][1]=n0+16; }
      else if (w > tv[i][2]) { tv[i][2]=w; ti[i][2]=n0+16; }
    }
  }

  // ---- phase 4: merge (ring idle: reuse LDS for candidates) ----
  __syncthreads();
#pragma unroll
  for (int i = 0; i < 8; ++i)
#pragma unroll
    for (int k = 0; k < KTOP; ++k) {
      sh.m.cval[wave][lane][i][k] = tv[i][k];
      sh.m.cidx[wave][lane][i][k] = ti[i][k];
    }
  __syncthreads();

  if (lane < 16) {
    const int slot = lane & 7;
    const int src0 = (lane < 8) ? 0 : 16;
    float v0 = -3.4e38f, v1 = v0, v2 = v0;
    int   j0 = 0, j1 = 0, j2 = 0;
    for (int s = 0; s < 16; ++s) {
#pragma unroll
      for (int k = 0; k < KTOP; ++k) {
        float v = sh.m.cval[wave][src0 + s][slot][k];
        int   j = sh.m.cidx[wave][src0 + s][slot][k];
        if (v > v0)      { v2=v1; j2=j1; v1=v0; j1=j0; v0=v; j0=j; }
        else if (v > v1) { v2=v1; j2=j1; v1=v;  j1=j; }
        else if (v > v2) { v2=v;  j2=j; }
      }
    }
    const int r = rowBase + (wave << 4) + lane;
    topv[r*KTOP+0] = v0; topv[r*KTOP+1] = v1; topv[r*KTOP+2] = v2;
    topi[r*KTOP+0] = j0; topi[r*KTOP+1] = j1; topi[r*KTOP+2] = j2;
    live[j0] = 1; live[j1] = 1; live[j2] = 1;
  }
}

// ---------------- fallback (small ws): inline-converted panels ----------------
static __device__ __forceinline__ void load_bpanel_fb(
    const float* __restrict__ Wenc, unsigned short* __restrict__ dst,
    int n0, int tid) {
  for (int e = tid * 4; e < 16 * MODEL_DIM; e += TPB * 4) {
    int n = e / MODEL_DIM;
    int k = e - n * MODEL_DIM;
    const float4 f = *reinterpret_cast<const float4*>(
        Wenc + (size_t)(n0 + n) * MODEL_DIM + k);
    unsigned long long p = (unsigned long long)f2bf(f.x)
                         | ((unsigned long long)f2bf(f.y) << 16)
                         | ((unsigned long long)f2bf(f.z) << 32)
                         | ((unsigned long long)f2bf(f.w) << 48);
    *reinterpret_cast<unsigned long long*>(dst + n * ASTRIDE + k) = p;
  }
}

__global__ __launch_bounds__(TPB, 1)
void sae_encode_topk_fb(const float* __restrict__ X,
                        const float* __restrict__ Wenc,
                        const float* __restrict__ benc,
                        float* __restrict__ topv, int* __restrict__ topi,
                        int* __restrict__ live)
{
  __shared__ alignas(16) unsigned short Alds[MB * ASTRIDE];
  __shared__ alignas(16) unsigned short Blds[2][16 * ASTRIDE];
  __shared__ float cval[WAVES][32][8][KTOP];
  __shared__ int   cidx[WAVES][32][8][KTOP];

  const int tid     = threadIdx.x;
  const int wave    = tid >> 5;
  const int lane    = tid & 31;
  const int rowBase = blockIdx.x * MB;

  for (int e = tid * 4; e < MB * MODEL_DIM; e += TPB * 4) {
    int m = e / MODEL_DIM;
    int k = e - m * MODEL_DIM;
    const float4 f = *reinterpret_cast<const float4*>(
        X + (size_t)(rowBase + m) * MODEL_DIM + k);
    unsigned long long p = (unsigned long long)f2bf(f.x)
                         | ((unsigned long long)f2bf(f.y) << 16)
                         | ((unsigned long long)f2bf(f.z) << 32)
                         | ((unsigned long long)f2bf(f.w) << 48);
    *reinterpret_cast<unsigned long long*>(&Alds[m * ASTRIDE + k]) = p;
  }

  float tv[8][KTOP]; int ti[8][KTOP];
#pragma unroll
  for (int i = 0; i < 8; ++i) {
    tv[i][0] = tv[i][1] = tv[i][2] = -3.4e38f;
    ti[i][0] = ti[i][1] = ti[i][2] = 0;
  }

  const int half = lane >> 4;
  const unsigned short* arow = &Alds[((wave << 4) + (lane & 15)) * ASTRIDE];
  const int bcol = lane & 15;

  load_bpanel_fb(Wenc, &Blds[0][0], 0, tid);

  const int NT = SAE_HIDDEN / 16;
  for (int t = 0; t < NT; ++t) {
    __syncthreads();
    if (t + 1 < NT)
      load_bpanel_fb(Wenc, &Blds[(t + 1) & 1][0], (t + 1) * 16, tid);
    if (t + 2 < NT && lane == 0)
      __builtin_prefetch(Wenc + (size_t)(t + 2) * 16 * MODEL_DIM, 0, 1);

    const unsigned short* bp = &Blds[t & 1][bcol * ASTRIDE];
    v8f c = {0.f,0.f,0.f,0.f,0.f,0.f,0.f,0.f};
#pragma unroll
    for (int k0 = 0; k0 < MODEL_DIM; k0 += 32) {
      v16bf a = ld_frag(arow + k0 + (half ? 8 : 0),
                        arow + k0 + 16 + (half ? 8 : 0));
      v16bf b = ld_frag(bp + k0 + (half ? 8 : 0),
                        bp + k0 + (half ? 24 : 16));
      c = __builtin_amdgcn_wmma_f32_16x16x32_bf16(
              false, a, false, b, (short)0, c, false, false);
    }
    const int ncol  = t * 16 + bcol;
    const float bia = benc[ncol];
#pragma unroll
    for (int i = 0; i < 8; ++i) {
      float v = c[i] + bia;
      if (v > tv[i][0]) { tv[i][2]=tv[i][1]; ti[i][2]=ti[i][1];
                          tv[i][1]=tv[i][0]; ti[i][1]=ti[i][0];
                          tv[i][0]=v;        ti[i][0]=ncol; }
      else if (v > tv[i][1]) { tv[i][2]=tv[i][1]; ti[i][2]=ti[i][1];
                               tv[i][1]=v; ti[i][1]=ncol; }
      else if (v > tv[i][2]) { tv[i][2]=v; ti[i][2]=ncol; }
    }
  }

  __syncthreads();
#pragma unroll
  for (int i = 0; i < 8; ++i)
#pragma unroll
    for (int k = 0; k < KTOP; ++k) {
      cval[wave][lane][i][k] = tv[i][k];
      cidx[wave][lane][i][k] = ti[i][k];
    }
  __syncthreads();

  if (lane < 16) {
    const int slot = lane & 7;
    const int src0 = (lane < 8) ? 0 : 16;
    float v0 = -3.4e38f, v1 = v0, v2 = v0;
    int   j0 = 0, j1 = 0, j2 = 0;
    for (int s = 0; s < 16; ++s) {
#pragma unroll
      for (int k = 0; k < KTOP; ++k) {
        float v = cval[wave][src0 + s][slot][k];
        int   j = cidx[wave][src0 + s][slot][k];
        if (v > v0)      { v2=v1; j2=j1; v1=v0; j1=j0; v0=v; j0=j; }
        else if (v > v1) { v2=v1; j2=j1; v1=v;  j1=j; }
        else if (v > v2) { v2=v;  j2=j; }
      }
    }
    const int r = rowBase + (wave << 4) + lane;
    topv[r*KTOP+0] = v0; topv[r*KTOP+1] = v1; topv[r*KTOP+2] = v2;
    topi[r*KTOP+0] = j0; topi[r*KTOP+1] = j1; topi[r*KTOP+2] = j2;
    live[j0] = 1; live[j1] = 1; live[j2] = 1;
  }
}

__global__ void sae_count_live(const int* __restrict__ live,
                               float* __restrict__ out_count) {
  __shared__ int red[256];
  int s = 0;
  for (int i = threadIdx.x; i < SAE_HIDDEN; i += 256) s += live[i];
  red[threadIdx.x] = s;
  __syncthreads();
  for (int off = 128; off > 0; off >>= 1) {
    if (threadIdx.x < (unsigned)off) red[threadIdx.x] += red[threadIdx.x + off];
    __syncthreads();
  }
  if (threadIdx.x == 0) *out_count = (float)red[0];
}

__global__ __launch_bounds__(256)
void sae_decode(const float* __restrict__ topv, const int* __restrict__ topi,
                const float* __restrict__ Wdec,   // [D, H] (L2-resident)
                const float* __restrict__ bdec,   // [D]
                float* __restrict__ out)          // [B, D]
{
  const int b  = blockIdx.x;
  const float v0 = topv[b*KTOP+0], v1 = topv[b*KTOP+1], v2 = topv[b*KTOP+2];
  const int   i0 = topi[b*KTOP+0], i1 = topi[b*KTOP+1], i2 = topi[b*KTOP+2];
  for (int d = threadIdx.x; d < MODEL_DIM; d += 256) {
    const float* wr = Wdec + (size_t)d * SAE_HIDDEN;
    out[(size_t)b * MODEL_DIM + d] =
        bdec[d] + v0 * wr[i0] + v1 * wr[i1] + v2 * wr[i2];
  }
}

extern "C" void kernel_launch(void* const* d_in, const int* in_sizes, int n_in,
                              void* d_out, int out_size, void* d_ws, size_t ws_size,
                              hipStream_t stream) {
  const float* X    = (const float*)d_in[0];
  const float* Wenc = (const float*)d_in[1];
  const float* benc = (const float*)d_in[2];
  const float* Wdec = (const float*)d_in[3];
  const float* bdec = (const float*)d_in[4];
  float* out = (float*)d_out;

  char*  ws   = (char*)d_ws;
  float* topv = (float*)ws;
  int*   topi = (int*)(ws + (size_t)BATCH * KTOP * 4);
  int*   live = (int*)(ws + (size_t)2 * BATCH * KTOP * 4);
  const size_t fixed = (size_t)2 * BATCH * KTOP * 4 + (size_t)SAE_HIDDEN * 4;
  unsigned short* WencBf = (unsigned short*)(ws + fixed);
  const size_t need = fixed + (size_t)SAE_HIDDEN * MODEL_DIM * 2;

  sae_zero_live<<<SAE_HIDDEN / 256, 256, 0, stream>>>(live);

  if (ws_size >= need) {
    const int n = SAE_HIDDEN * MODEL_DIM;
    sae_cvt_bf16<<<n / 1024, 256, 0, stream>>>(Wenc, WencBf, n);
    sae_encode_topk_tdm<<<BATCH / MB, TPB, 0, stream>>>(
        X, WencBf, benc, topv, topi, live);
  } else {
    sae_encode_topk_fb<<<BATCH / MB, TPB, 0, stream>>>(
        X, Wenc, benc, topv, topi, live);
  }

  sae_count_live<<<1, 256, 0, stream>>>(live, out + (size_t)BATCH * MODEL_DIM);
  sae_decode<<<BATCH, 256, 0, stream>>>(topv, topi, Wdec, bdec, out);
}
